// Attention_Encoder_10797547782168
// MI455X (gfx1250) — matrix operand order repaired and checked by reading, not compiled, not run on hardware
//
#include <hip/hip_runtime.h>
#include <hip/hip_bf16.h>
#include <stddef.h>
#include <stdint.h>

// ---------------------------------------------------------------------------
// Problem constants (from reference)
// ---------------------------------------------------------------------------
#define NSEG 1024      // B*LTMS
#define TTS  256
#define C_IN 128
#define FDIM 256
#define HDIM 256
#define G3H  768       // 3*H

// CDNA5 wave32 WMMA types
typedef __bf16 bf16_t;
typedef __attribute__((ext_vector_type(16))) __bf16 v16bf;
typedef __attribute__((ext_vector_type(8)))  float  v8f;

union ABf {
    v16bf v;
    unsigned short us[16];
    unsigned int   u[8];
};

// ---- gfx1250 async Global<->LDS path (ASYNCcnt), guarded -------------------
#if __has_builtin(__builtin_amdgcn_global_load_async_to_lds_b128) && \
    __has_builtin(__builtin_amdgcn_s_wait_asynccnt)
#define HAVE_ASYNC_LD 1
#else
#define HAVE_ASYNC_LD 0
#endif
#if __has_builtin(__builtin_amdgcn_global_store_async_from_lds_b128) && \
    __has_builtin(__builtin_amdgcn_s_wait_asynccnt)
#define HAVE_ASYNC_ST 1
#else
#define HAVE_ASYNC_ST 0
#endif

#if HAVE_ASYNC_LD || HAVE_ASYNC_ST
// builtin prototype (from hipcc diagnostic): (v4i AS1*, v4i AS3*, imm, imm)
typedef int v4i_async __attribute__((vector_size(16)));
typedef __attribute__((address_space(1))) v4i_async gv4i;  // global
typedef __attribute__((address_space(3))) v4i_async lv4i;  // LDS
#endif

#if HAVE_ASYNC_LD
__device__ __forceinline__ void async_ld16(void* lds, const void* g) {
    __builtin_amdgcn_global_load_async_to_lds_b128(
        (gv4i*)(uintptr_t)g,
        (lv4i*)(unsigned int)(uintptr_t)lds,   // low 32 bits = LDS offset (aperture rule)
        0, 0);
}
#endif
#if HAVE_ASYNC_ST
__device__ __forceinline__ void async_st16(void* g, const void* lds) {
    __builtin_amdgcn_global_store_async_from_lds_b128(
        (gv4i*)(uintptr_t)g,
        (lv4i*)(unsigned int)(uintptr_t)lds,
        0, 0);
}
#endif

// ---- gfx1250 Tensor Data Mover (TENSORcnt), guarded ------------------------
#if __has_builtin(__builtin_amdgcn_tensor_load_to_lds) && \
    __has_builtin(__builtin_amdgcn_s_wait_tensorcnt)
#define HAVE_TDM 1
#else
#define HAVE_TDM 0
#endif

#if HAVE_TDM
typedef unsigned int u32x4_tdm __attribute__((ext_vector_type(4)));
typedef int          i32x4_tdm __attribute__((ext_vector_type(4)));
typedef int          i32x8_tdm __attribute__((ext_vector_type(8)));

// 2D TDM tile load: tile_d0 x tile_d1 elements (data_size bytes each) from a
// strided 2D tensor into LDS, with optional LDS row padding (ISA 8.3/8.4).
__device__ __forceinline__ void tdm_load_2d(unsigned int lds_byte_addr,
                                            const void* gaddr,
                                            unsigned int g1_dw0,        // flags: data_size/pad
                                            unsigned int tensor_d0,
                                            unsigned int tensor_d1,
                                            unsigned int tile_d0,
                                            unsigned int tile_d1,
                                            unsigned long long d0_stride_elems) {
    unsigned long long ga = (unsigned long long)(uintptr_t)gaddr;
    u32x4_tdm g0;
    g0[0] = 1u;                                              // count=1 valid descriptor
    g0[1] = lds_byte_addr;                                   // lds_addr
    g0[2] = (unsigned int)ga;                                // global_addr[31:0]
    g0[3] = (unsigned int)((ga >> 32) & 0x1FFFFFFu)          // global_addr[56:32]
          | (2u << 30);                                      // type=2 ("image")
    i32x8_tdm g1;
    g1[0] = (int)g1_dw0;
    g1[1] = (int)((tensor_d0 & 0xFFFFu) << 16);              // abar_addr=0 | tensor_dim0[15:0]
    g1[2] = (int)(((tensor_d0 >> 16) & 0xFFFFu) | ((tensor_d1 & 0xFFFFu) << 16));
    g1[3] = (int)(((tensor_d1 >> 16) & 0xFFFFu) | ((tile_d0 & 0xFFFFu) << 16));
    g1[4] = (int)(tile_d1 & 0xFFFFu);                        // tile_dim1 | tile_dim2=0
    g1[5] = (int)(unsigned int)(d0_stride_elems & 0xFFFFFFFFu);
    g1[6] = (int)(unsigned int)((d0_stride_elems >> 32) & 0xFFFFu);  // stride[47:32] | dim1_stride lo = 0
    g1[7] = 0;
    i32x4_tdm z4 = {0, 0, 0, 0};
#if __clang_major__ >= 23
    i32x8_tdm z8 = {0, 0, 0, 0, 0, 0, 0, 0};
    __builtin_amdgcn_tensor_load_to_lds(g0, g1, z4, z4, z8, 0);
#else
    __builtin_amdgcn_tensor_load_to_lds(g0, g1, z4, z4, 0);
#endif
}
#endif

__device__ __forceinline__ unsigned short f32_to_bf16(float f) {
    unsigned int u = __float_as_uint(f);
    u += 0x7FFFu + ((u >> 16) & 1u);   // round-to-nearest-even
    return (unsigned short)(u >> 16);
}
__device__ __forceinline__ float bf16u_to_f32(unsigned short h) {
    return __uint_as_float(((unsigned int)h) << 16);
}

__device__ __forceinline__ v8f wmma_bf16(v16bf a, v16bf b, v8f c) {
    // D(16x16 f32) = A(16x32 bf16) * B(32x16 bf16) + C
    return __builtin_amdgcn_wmma_f32_16x16x32_bf16(
        false, a, false, b, (short)0, c, false, false);
}

// Packed B layout: for (ntile, kk) block, 32 lanes x 16 bf16 contiguous.
// lane l holds column n = ntile*16 + (l%16), K = kk*32 + (l/16)*8 + (i%8) + (i/8)*16.
__device__ __forceinline__ v16bf load_b_pack(const unsigned short* __restrict__ pack,
                                             int blk, int lane) {
    const uint4* p = (const uint4*)(pack + (((size_t)blk * 32 + lane) << 4));
    ABf b;
    *(uint4*)&b.u[0] = p[0];
    *(uint4*)&b.u[4] = p[1];
    return b.v;
}

// A-tile gather from an LDS row (row-major, K contiguous): pairs are dword loads.
__device__ __forceinline__ void gather_a(const unsigned short* __restrict__ row,
                                         int kb, ABf& A) {
#pragma unroll
    for (int p = 0; p < 4; ++p) {
        A.u[p]     = *(const unsigned int*)&row[kb + 2 * p];
        A.u[p + 4] = *(const unsigned int*)&row[kb + 16 + 2 * p];
    }
}

// ---------------------------------------------------------------------------
// Kernel 0: pack an fp32 [K][N] weight matrix into the WMMA-B bf16 layout.
// ---------------------------------------------------------------------------
__global__ void pack_b_kernel(const float* __restrict__ src,
                              unsigned short* __restrict__ dst,
                              int K, int N) {
    int total = K * N;
    int nkk = K >> 5;
    for (int e = blockIdx.x * blockDim.x + threadIdx.x; e < total;
         e += gridDim.x * blockDim.x) {
        int i    = e & 15;
        int lane = (e >> 4) & 31;
        int blk  = e >> 9;
        int kk   = blk % nkk;
        int nt   = blk / nkk;
        int k = kk * 32 + (lane >> 4) * 8 + (i & 7) + (i >> 3) * 16;
        int n = nt * 16 + (lane & 15);
        dst[e] = f32_to_bf16(src[(size_t)k * N + n]);
    }
}

// ---------------------------------------------------------------------------
// Kernel 1: per-segment  conv(1x1)+ReLU -> attention softmax -> x_att (bf16)
// One workgroup (8 wave32) per segment. ~200KB dynamic LDS.
// ---------------------------------------------------------------------------
#define XS_STRIDE 136   // 128 + 8 pad (ushort)
#define CV_STRIDE 264   // 256 + 8 pad (ushort)

__global__ void __launch_bounds__(256)
seg_kernel(const float* __restrict__ x,
           const float* __restrict__ conv_b,
           const float* __restrict__ attn_b,
           const unsigned short* __restrict__ wc_pack,
           const unsigned short* __restrict__ attn_pack,
           unsigned short* __restrict__ xatt) {
    extern __shared__ char smem[];
    unsigned short* xs_lds = (unsigned short*)smem;                         // [256][136]
    unsigned short* cv_lds = (unsigned short*)(smem + TTS * XS_STRIDE * 2); // [256][264]

    const int tid  = threadIdx.x;
    const int wave = tid >> 5;
    const int lane = tid & 31;
    const int lm   = lane & 15;
    const int lh   = lane >> 4;
    const int seg  = blockIdx.x;

    // ---- stage x segment as bf16 into LDS (fp32->bf16 conversion in VALU) ----
    const float* xseg = x + (size_t)seg * TTS * C_IN;
    for (int idx = tid; idx < TTS * C_IN; idx += 256) {
        int t = idx >> 7, c = idx & 127;
        xs_lds[t * XS_STRIDE + c] = f32_to_bf16(xseg[idx]);
    }
    __syncthreads();

    // ---- GEMM1: conv = relu(x[256x128] @ Wc[128x256] + b) -> cv_lds bf16 ----
    for (int tr = wave * 2; tr < wave * 2 + 2; ++tr) {
        ABf A[4];
        const unsigned short* row = &xs_lds[(tr * 16 + lm) * XS_STRIDE];
#pragma unroll
        for (int kk = 0; kk < 4; ++kk) gather_a(row, kk * 32 + lh * 8, A[kk]);

        for (int ft = 0; ft < 16; ++ft) {
            v8f acc = {};
#pragma unroll
            for (int kk = 0; kk < 4; ++kk)
                acc = wmma_bf16(A[kk].v, load_b_pack(wc_pack, ft * 4 + kk, lane), acc);
            float bias = conv_b[ft * 16 + lm];
#pragma unroll
            for (int v = 0; v < 8; ++v) {
                float val = acc[v] + bias;
                val = val > 0.f ? val : 0.f;
                int t = tr * 16 + v + lh * 8;           // C layout: M = v + 8*(lane/16)
                cv_lds[t * CV_STRIDE + ft * 16 + lm] = f32_to_bf16(val);
            }
        }
    }
    __syncthreads();

    // ---- GEMM2 + softmax + in-place x_att. Each wave owns f-blocks fb, fb+8 ----
    for (int fb = wave; fb < 16; fb += 8) {
        v8f acc[16];
#pragma unroll
        for (int j = 0; j < 16; ++j) acc[j] = (v8f){};

        const int f = fb * 16 + lm;   // A row (f') = lane%16; K axis = t
        for (int kk = 0; kk < 8; ++kk) {
            ABf A;
            int tb = kk * 32 + lh * 8;
#pragma unroll
            for (int p = 0; p < 4; ++p) {
                unsigned int lo = cv_lds[(tb + 2 * p) * CV_STRIDE + f];
                unsigned int hi = cv_lds[(tb + 2 * p + 1) * CV_STRIDE + f];
                A.u[p] = lo | (hi << 16);
                lo = cv_lds[(tb + 16 + 2 * p) * CV_STRIDE + f];
                hi = cv_lds[(tb + 16 + 2 * p + 1) * CV_STRIDE + f];
                A.u[p + 4] = lo | (hi << 16);
            }
#pragma unroll
            for (int j = 0; j < 16; ++j)
                acc[j] = wmma_bf16(A.v, load_b_pack(attn_pack, j * 8 + kk, lane), acc[j]);
        }
#pragma unroll
        for (int j = 0; j < 16; ++j) {
            float ab = attn_b[j * 16 + lm];
#pragma unroll
            for (int v = 0; v < 8; ++v) acc[j][v] += ab;
        }
        // softmax over t' per f-row: in-lane over tiles j, then 16-lane-half shfls
#pragma unroll
        for (int v = 0; v < 8; ++v) {
            float m = -3.0e38f;
#pragma unroll
            for (int j = 0; j < 16; ++j) m = fmaxf(m, acc[j][v]);
            m = fmaxf(m, __shfl_xor(m, 1, 32));
            m = fmaxf(m, __shfl_xor(m, 2, 32));
            m = fmaxf(m, __shfl_xor(m, 4, 32));
            m = fmaxf(m, __shfl_xor(m, 8, 32));
            float s = 0.f;
#pragma unroll
            for (int j = 0; j < 16; ++j) {
                float e = __expf(acc[j][v] - m);
                acc[j][v] = e;
                s += e;
            }
            s += __shfl_xor(s, 1, 32);
            s += __shfl_xor(s, 2, 32);
            s += __shfl_xor(s, 4, 32);
            s += __shfl_xor(s, 8, 32);
            float inv = 1.f / s;
#pragma unroll
            for (int j = 0; j < 16; ++j) acc[j][v] *= inv;
        }
        // x_att[t', f] = conv[t', f] * probs[f, t']  (wave-private f columns)
#pragma unroll
        for (int j = 0; j < 16; ++j) {
#pragma unroll
            for (int v = 0; v < 8; ++v) {
                int ff = fb * 16 + v + lh * 8;
                int tt = j * 16 + lm;
                float cvv = bf16u_to_f32(cv_lds[tt * CV_STRIDE + ff]);
                cv_lds[tt * CV_STRIDE + ff] = f32_to_bf16(cvv * acc[j][v]);
            }
        }
    }
    __syncthreads();

    // ---- store x_att bf16 to global ----
    unsigned short* xo = xatt + (size_t)seg * TTS * FDIM;
#if HAVE_ASYNC_ST
    // LDS -> global via ASYNCcnt-tracked b128 DMA (no VGPR round trip).
    for (int c = tid; c < TTS * 32; c += 256) {     // 256 rows x 32 16B chunks
        int t  = c >> 5;
        int e8 = (c & 31) * 8;
        async_st16(xo + (size_t)t * FDIM + e8, &cv_lds[t * CV_STRIDE + e8]);
    }
    __builtin_amdgcn_s_wait_asynccnt(0);
#else
    unsigned int* xout = (unsigned int*)xo;
    for (int idx = tid; idx < (TTS * FDIM) / 2; idx += 256) {
        int t  = idx >> 7;
        int f2 = (idx & 127) << 1;
        unsigned int lo = cv_lds[t * CV_STRIDE + f2];
        unsigned int hi = cv_lds[t * CV_STRIDE + f2 + 1];
        xout[idx] = lo | (hi << 16);
    }
#endif
}

// ---------------------------------------------------------------------------
// Kernel 2: GRU scan. 32 WGs x 32 segments each: recurrence is an M=32 WMMA
// GEMM per step; each B-tile load feeds two M-tiles (2x arithmetic intensity
// on the L2-resident weights). gates_x GEMM fused into the scan. Per-step
// x_att tile is staged by the Tensor Data Mover (hardware row padding to
// HR_STRIDE), falling back to async-LDS loads / plain loads.
// ---------------------------------------------------------------------------
#define HR_STRIDE 264
#define SEGS 32

__global__ void __launch_bounds__(256)
gru_kernel(const unsigned short* __restrict__ xatt,
           const unsigned short* __restrict__ gruw_pack,
           const unsigned short* __restrict__ gruu_pack,
           const float* __restrict__ gru_b,
           float* __restrict__ out) {
    extern __shared__ char gsm[];
    unsigned short* xr_lds = (unsigned short*)gsm;                            // [32][264] bf16
    unsigned short* hb_lds = (unsigned short*)(gsm + SEGS * HR_STRIDE * 2);   // [32][264] bf16
    float*          hf_lds = (float*)(gsm + SEGS * HR_STRIDE * 4);            // [32][264] f32

    const int tid  = threadIdx.x;
    const int wave = tid >> 5;
    const int lane = tid & 31;
    const int lm   = lane & 15;
    const int lh   = lane >> 4;
    const int seg0 = blockIdx.x * SEGS;

    const float* bi  = gru_b;          // gru_b[0]
    const float* brv = gru_b + G3H;    // gru_b[1]

    for (int i = tid; i < SEGS * HR_STRIDE; i += 256) {
        hb_lds[i] = 0;
        hf_lds[i] = 0.f;
    }
    __syncthreads();

    for (int t = 0; t < TTS; ++t) {
        // ---- stage x_att[seg0..seg0+31, t, :] into LDS ----
#if HAVE_TDM
        if (wave == 0) {
            // 2D tile: 256 elems (data_size=2B) x 32 rows; row pitch = TTS*FDIM
            // elems; LDS pad: after every 128 DWORDs (512B) add 4 DWORDs (16B)
            // => hardware reproduces HR_STRIDE=264.
            const unsigned int g1_dw0 = (1u << 16)   // data_size = 2 bytes
                                      | (1u << 20)   // pad_enable
                                      | (6u << 22)   // pad_interval: 128 DWORDs
                                      | (3u << 25);  // pad_amount: 4 DWORDs
            tdm_load_2d((unsigned int)(uintptr_t)xr_lds,
                        xatt + (size_t)seg0 * (TTS * FDIM) + (size_t)t * FDIM,
                        g1_dw0,
                        /*tensor_d0=*/FDIM, /*tensor_d1=*/SEGS,
                        /*tile_d0=*/FDIM,   /*tile_d1=*/SEGS,
                        /*d0_stride=*/(unsigned long long)TTS * FDIM);
            __builtin_amdgcn_s_wait_tensorcnt(0);
        }
#elif HAVE_ASYNC_LD
        for (int c = tid; c < SEGS * 32; c += 256) {  // 32 rows x 32 16B chunks
            int m  = c >> 5;
            int e8 = (c & 31) * 8;
            const unsigned short* src = xatt + (size_t)(seg0 + m) * (TTS * FDIM)
                                             + (size_t)t * FDIM + e8;
            async_ld16(&xr_lds[m * HR_STRIDE + e8], src);
        }
        __builtin_amdgcn_s_wait_asynccnt(0);
#else
        for (int idx = tid; idx < (SEGS * FDIM) / 2; idx += 256) {
            int m  = idx >> 7;
            int f2 = (idx & 127) << 1;
            const unsigned int* src = (const unsigned int*)
                (xatt + (size_t)(seg0 + m) * (TTS * FDIM) + (size_t)t * FDIM);
            *(unsigned int*)&xr_lds[m * HR_STRIDE + f2] = src[idx & 127];
        }
#endif
        // prefetch next step's rows while this step computes
        if (t + 1 < TTS) {
            __builtin_prefetch(xatt + (size_t)(seg0 + (tid & (SEGS - 1))) * (TTS * FDIM)
                                    + (size_t)(t + 1) * FDIM, 0, 1);
        }
        __syncthreads();

        // ---- gx = X @ gru_w ; rec = h @ gru_u.  wave: 2 hidden tiles x 3 gates x 2 M-tiles
        v8f gx[3][2][2];
        v8f rc[3][2][2];
#pragma unroll
        for (int g = 0; g < 3; ++g)
#pragma unroll
            for (int jj = 0; jj < 2; ++jj)
#pragma unroll
                for (int mt = 0; mt < 2; ++mt) { gx[g][jj][mt] = (v8f){}; rc[g][jj][mt] = (v8f){}; }

        for (int kk = 0; kk < 8; ++kk) {
            ABf Ax[2], Ah[2];
            int kb = kk * 32 + lh * 8;
#pragma unroll
            for (int mt = 0; mt < 2; ++mt) {
                gather_a(&xr_lds[(mt * 16 + lm) * HR_STRIDE], kb, Ax[mt]);
                gather_a(&hb_lds[(mt * 16 + lm) * HR_STRIDE], kb, Ah[mt]);
            }
#pragma unroll
            for (int g = 0; g < 3; ++g) {
#pragma unroll
                for (int jj = 0; jj < 2; ++jj) {
                    int nt = g * 16 + wave * 2 + jj;   // gate g: N cols [g*256, g*256+256)
                    v16bf bw = load_b_pack(gruw_pack, nt * 8 + kk, lane);
                    v16bf bu = load_b_pack(gruu_pack, nt * 8 + kk, lane);
#pragma unroll
                    for (int mt = 0; mt < 2; ++mt) {
                        gx[g][jj][mt] = wmma_bf16(Ax[mt].v, bw, gx[g][jj][mt]);
                        rc[g][jj][mt] = wmma_bf16(Ah[mt].v, bu, rc[g][jj][mt]);
                    }
                }
            }
        }
        __syncthreads();   // all reads of hb_lds done before update

        // ---- combine gates (keras reset_after GRU, activation=relu, order z,r,h)
#pragma unroll
        for (int jj = 0; jj < 2; ++jj) {
            int n = (wave * 2 + jj) * 16 + lm;
            float biz = bi[n],  bir = bi[HDIM + n],  bih = bi[2 * HDIM + n];
            float brz = brv[n], brr = brv[HDIM + n], brh = brv[2 * HDIM + n];
#pragma unroll
            for (int mt = 0; mt < 2; ++mt) {
#pragma unroll
                for (int v = 0; v < 8; ++v) {
                    int m = mt * 16 + v + lh * 8;
                    float xz = gx[0][jj][mt][v] + biz;
                    float xr = gx[1][jj][mt][v] + bir;
                    float xh = gx[2][jj][mt][v] + bih;
                    float rz = rc[0][jj][mt][v] + brz;
                    float rr = rc[1][jj][mt][v] + brr;
                    float rh = rc[2][jj][mt][v] + brh;
                    float z = 1.f / (1.f + __expf(-(xz + rz)));
                    float r = 1.f / (1.f + __expf(-(xr + rr)));
                    float hh = xh + r * rh;
                    hh = hh > 0.f ? hh : 0.f;
                    float hold = hf_lds[m * HR_STRIDE + n];
                    float hnew = z * hold + (1.f - z) * hh;
                    hf_lds[m * HR_STRIDE + n] = hnew;
                    hb_lds[m * HR_STRIDE + n] = f32_to_bf16(hnew);
                }
            }
        }
        __syncthreads();   // h updated before next step reads it
    }

    // final hidden state -> d_out [NSEG][H] fp32
    for (int idx = tid; idx < SEGS * HDIM; idx += 256) {
        int m = idx >> 8, n = idx & 255;
        out[(size_t)(seg0 + m) * HDIM + n] = hf_lds[m * HR_STRIDE + n];
    }
}

// ---------------------------------------------------------------------------
// Launch
// ---------------------------------------------------------------------------
extern "C" void kernel_launch(void* const* d_in, const int* in_sizes, int n_in,
                              void* d_out, int out_size, void* d_ws, size_t ws_size,
                              hipStream_t stream) {
    (void)in_sizes; (void)n_in; (void)out_size; (void)ws_size;

    const float* x      = (const float*)d_in[0];
    const float* conv_w = (const float*)d_in[1];  // [1][128][256]
    const float* conv_b = (const float*)d_in[2];
    const float* attn_w = (const float*)d_in[3];  // [256][256]
    const float* attn_b = (const float*)d_in[4];
    const float* gru_w  = (const float*)d_in[5];  // [256][768]
    const float* gru_u  = (const float*)d_in[6];  // [256][768]
    const float* gru_b  = (const float*)d_in[7];  // [2][768]

    char* ws = (char*)d_ws;
    unsigned short* wc_pack   = (unsigned short*)(ws + 0);        //  64 KB
    unsigned short* attn_pack = (unsigned short*)(ws + 65536);    // 128 KB
    unsigned short* gruw_pack = (unsigned short*)(ws + 196608);   // 384 KB
    unsigned short* gruu_pack = (unsigned short*)(ws + 589824);   // 384 KB
    unsigned short* xatt      = (unsigned short*)(ws + 983040);   // 128 MB bf16

    pack_b_kernel<<<64,  256, 0, stream>>>(conv_w, wc_pack,   C_IN, FDIM);
    pack_b_kernel<<<128, 256, 0, stream>>>(attn_w, attn_pack, TTS,  TTS);
    pack_b_kernel<<<384, 256, 0, stream>>>(gru_w,  gruw_pack, FDIM, G3H);
    pack_b_kernel<<<384, 256, 0, stream>>>(gru_u,  gruu_pack, HDIM, G3H);

    size_t seg_lds = (size_t)(TTS * XS_STRIDE + TTS * CV_STRIDE) * 2; // ~200 KB dynamic LDS
    seg_kernel<<<NSEG, 256, seg_lds, stream>>>(x, conv_b, attn_b,
                                               wc_pack, attn_pack, xatt);

    size_t gru_lds = (size_t)SEGS * HR_STRIDE * 8;                    // ~66 KB dynamic LDS
    gru_kernel<<<NSEG / SEGS, 256, gru_lds, stream>>>(xatt, gruw_pack, gruu_pack,
                                                      gru_b, (float*)d_out);
}
